// PointUNet_75024488727154
// MI455X (gfx1250) — matrix operand order repaired and checked by reading, not compiled, not run on hardware
//
#include <hip/hip_runtime.h>
#include <hip/hip_bf16.h>

typedef __attribute__((ext_vector_type(16))) _Float16 v16h;
typedef __attribute__((ext_vector_type(8)))  _Float16 v8h;
typedef __attribute__((ext_vector_type(8)))  float    v8f;

#define WPB 4   // waves per block in WMMA kernels

__device__ __forceinline__ float leakyf(float v) { return v >= 0.f ? v : 0.2f * v; }

// ---------------------------------------------------------------------------
// Weight conversion: w [cin, cout] f32 row-major -> wt [cout, kpad] f16
// (transposed, zero-padded K) so B-fragments are contiguous 16-half loads.
// ---------------------------------------------------------------------------
__global__ void wt_convert_kernel(const float* __restrict__ w, _Float16* __restrict__ wt,
                                  int cin, int cout, int kpad) {
  int i = blockIdx.x * blockDim.x + threadIdx.x;
  if (i >= cout * kpad) return;
  int n = i / kpad, k = i - n * kpad;
  wt[(size_t)n * kpad + k] = (k < cin) ? (_Float16)w[(size_t)k * cout + n] : (_Float16)0.f;
}

// ---------------------------------------------------------------------------
// Brute-force 16-NN (includes self, ascending distance, stable on ties).
// One query per thread, candidates tiled through LDS, next tile prefetched.
// ---------------------------------------------------------------------------
__global__ void knn16_kernel(const float* __restrict__ pos, int* __restrict__ knn, int M) {
  int b = blockIdx.y;
  const float* P = pos + (size_t)b * M * 3;
  int q = blockIdx.x * blockDim.x + threadIdx.x;
  __shared__ float sx[256], sy[256], sz[256];
  float qx = 0.f, qy = 0.f, qz = 0.f;
  if (q < M) { qx = P[q*3+0]; qy = P[q*3+1]; qz = P[q*3+2]; }
  float bd[16]; int bi[16];
#pragma unroll
  for (int i = 0; i < 16; ++i) { bd[i] = 3.0e38f; bi[i] = 0; }
  for (int base = 0; base < M; base += 256) {
    int c = base + (int)threadIdx.x;
    if (c < M) { sx[threadIdx.x] = P[c*3+0]; sy[threadIdx.x] = P[c*3+1]; sz[threadIdx.x] = P[c*3+2]; }
    int pf = c + 256;
    if (pf < M) __builtin_prefetch(&P[(size_t)pf * 3], 0, 0);   // global_prefetch_b8
    __syncthreads();
    int lim = M - base; if (lim > 256) lim = 256;
    if (q < M) {
      for (int j = 0; j < lim; ++j) {
        float dx = qx - sx[j], dy = qy - sy[j], dz = qz - sz[j];
        float d = dx*dx + dy*dy + dz*dz;
        if (d < bd[15]) {
          float cd = d; int ci = base + j;
#pragma unroll
          for (int t = 0; t < 16; ++t) {
            if (cd < bd[t]) { float td = bd[t]; int ti = bi[t]; bd[t] = cd; bi[t] = ci; cd = td; ci = ti; }
          }
        }
      }
    }
    __syncthreads();
  }
  if (q < M) {
#pragma unroll
    for (int i = 0; i < 16; ++i) knn[((size_t)b*M + q)*16 + i] = bi[i];
  }
}

// ---------------------------------------------------------------------------
// Furthest point sampling: one persistent workgroup per batch. Each thread
// owns N/512 points (coords in registers), serial np-step argmax loop with
// LDS tree reduction (lowest-index tiebreak like jnp.argmax).
// ---------------------------------------------------------------------------
#define FPS_T 512
__global__ void fps_kernel(const float* __restrict__ pos, int* __restrict__ out_idx,
                           int N, int np) {
  int b = blockIdx.x;
  const float* P = pos + (size_t)b * N * 3;
  int* O = out_idx + (size_t)b * np;
  int tid = threadIdx.x;
  float px[16], py[16], pz[16], md[16];
  int nown = 0;
  for (int i = tid, s = 0; i < N; i += FPS_T, ++s) {
    px[s] = P[i*3+0]; py[s] = P[i*3+1]; pz[s] = P[i*3+2];
    md[s] = 1.0e10f; nown = s + 1;
  }
  __shared__ float sval[FPS_T];
  __shared__ int   sidx[FPS_T];
  __shared__ int   sfar;
  if (tid == 0) sfar = 0;
  __syncthreads();
  for (int it = 0; it < np; ++it) {
    int far = sfar;
    if (tid == 0) O[it] = far;
    float cx = P[far*3+0], cy = P[far*3+1], cz = P[far*3+2];
    float bv = -1.f; int bix = 0x7fffffff;
    for (int s = 0; s < nown; ++s) {
      float dx = px[s]-cx, dy = py[s]-cy, dz = pz[s]-cz;
      float d = dx*dx + dy*dy + dz*dz;
      float m = md[s] < d ? md[s] : d;
      md[s] = m;
      int gi = tid + s * FPS_T;
      if (m > bv) { bv = m; bix = gi; }
    }
    sval[tid] = bv; sidx[tid] = bix;
    __syncthreads();
    for (int off = FPS_T/2; off > 0; off >>= 1) {
      if (tid < off) {
        float ov = sval[tid+off]; int oi = sidx[tid+off];
        if (ov > sval[tid] || (ov == sval[tid] && oi < sidx[tid])) { sval[tid] = ov; sidx[tid] = oi; }
      }
      __syncthreads();
    }
    if (tid == 0) sfar = sidx[0];
    __syncthreads();
  }
}

// ---------------------------------------------------------------------------
// Row gather: dst[b,m,:] = src[b, idx[b,m], :]
// ---------------------------------------------------------------------------
__global__ void gather_kernel(const float* __restrict__ src, const int* __restrict__ idx,
                              float* __restrict__ dst, int Min, int Mout, int C) {
  int b = blockIdx.y;
  int i = blockIdx.x * blockDim.x + threadIdx.x;
  if (i >= Mout * C) return;
  int m = i / C, c = i - m * C;
  int s = idx[(size_t)b * Mout + m];
  dst[((size_t)b*Mout + m)*C + c] = src[((size_t)b*Min + s)*C + c];
}

// ---------------------------------------------------------------------------
// Fused edge conv: WPB waves per block, one point per wave. The point's 16
// neighbor rows h=[nx, rel, d2] (K=16 == WMMA M) are staged in a private LDS
// slice as f16, multiplied by the transposed edge weight via
// v_wmma_f32_16x16x32_f16; mean-over-K = 8-VGPR sum + shfl_xor(16); self
// term + leaky fused into the epilogue.
// ---------------------------------------------------------------------------
__global__ void __launch_bounds__(32 * WPB)
edgeconv_wmma_kernel(const float* __restrict__ feat, const float* __restrict__ pos,
                     const int* __restrict__ knn, const float* __restrict__ w_self,
                     const _Float16* __restrict__ wt_edge, float* __restrict__ out,
                     int M, int C, int Cout, int Kpad) {
  __shared__ __align__(32) _Float16 Ast[WPB][16 * 160];   // Kpad <= 160
  int lane = threadIdx.x & 31;
  int w    = threadIdx.x >> 5;
  int row = lane & 15, half = lane >> 4;
  int b = blockIdx.y;
  int m = blockIdx.x * WPB + w;
  bool valid = m < M;
  _Float16* As = &Ast[w][0];
  const float* P = pos + (size_t)b * M * 3;
  if (valid) {
    int j = knn[((size_t)b*M + m)*16 + row];
    float rx = P[m*3+0] - P[j*3+0];
    float ry = P[m*3+1] - P[j*3+1];
    float rz = P[m*3+2] - P[j*3+2];
    float d2 = rx*rx + ry*ry + rz*rz;
    const float* Fj = feat + ((size_t)b*M + j) * C;
    for (int k = half; k < Kpad; k += 2) {
      float v;
      if (k < C)            v = Fj[k];
      else if (k == C)      v = rx;
      else if (k == C + 1)  v = ry;
      else if (k == C + 2)  v = rz;
      else if (k == C + 3)  v = d2;
      else                  v = 0.f;
      As[row * Kpad + k] = (_Float16)v;
    }
  }
  __syncthreads();
  if (!valid) return;
  const float* Fm = feat + ((size_t)b*M + m) * C;
  int nct = Cout >> 4, nkt = Kpad >> 5;
  for (int ct = 0; ct < nct; ++ct) {
    v8f acc = {};
    for (int kt = 0; kt < nkt; ++kt) {
      int kb = kt*32 + half*16;
      v16h a = *(const v16h*)&As[row*Kpad + kb];
      const _Float16* wr = wt_edge + (size_t)(ct*16 + row) * Kpad + kt*32 + half*16;
      v16h bm = *(const v16h*)wr;
      acc = __builtin_amdgcn_wmma_f32_16x16x32_f16(false, a, false, bm, (short)0, acc, false, false);
    }
    // lane L holds rows {0..7 or 8..15} of output column (ct*16 + L&15):
    float s = acc[0]+acc[1]+acc[2]+acc[3]+acc[4]+acc[5]+acc[6]+acc[7];
    s += __shfl_xor(s, 16, 32);
    if (half == 0) {
      int ch = ct*16 + row;
      float sv = 0.f;
      for (int c = 0; c < C; ++c) sv += Fm[c] * w_self[(size_t)c * Cout + ch];
      out[((size_t)b*M + m)*Cout + ch] = leakyf(sv + s * (1.f/16.f));
    }
  }
}

// ---------------------------------------------------------------------------
// WMMA linear: Y[rows,Cout] = act(X[rows,Cin] @ W + bias). WPB waves per
// block, one 16-row tile per wave; X chunks staged f32->f16 in a private LDS
// slice; up to 8 C/D tiles live in registers.
// ---------------------------------------------------------------------------
__global__ void __launch_bounds__(32 * WPB)
linear_wmma_kernel(const float* __restrict__ X, const _Float16* __restrict__ Wt,
                   const float* __restrict__ bias, float* __restrict__ Y,
                   int rows, int Cin, int Cout, int Kpad, int act) {
  __shared__ __align__(32) _Float16 Ast[WPB][16 * 32];
  int lane = threadIdx.x & 31;
  int w    = threadIdx.x >> 5;
  int row = lane & 15, half = lane >> 4;
  int r0 = (blockIdx.x * WPB + w) * 16;
  int gr = r0 + row;
  _Float16* As = &Ast[w][0];
  const float* Xr = X + (size_t)gr * Cin;
  int nct = Cout >> 4, nkt = Kpad >> 5;
  v8f acc[8] = {};
  for (int kt = 0; kt < nkt; ++kt) {
    __syncthreads();
#pragma unroll
    for (int i = 0; i < 16; ++i) {
      int k = kt*32 + half*16 + i;
      float v = (gr < rows && k < Cin) ? Xr[k] : 0.f;
      As[row*32 + half*16 + i] = (_Float16)v;
    }
    __syncthreads();
    int kb = half*16;
    v16h a = *(const v16h*)&As[row*32 + kb];
#pragma unroll
    for (int ct = 0; ct < 8; ++ct) {
      if (ct < nct) {
        const _Float16* wr = Wt + (size_t)(ct*16 + row) * Kpad + kt*32 + half*16;
        v16h bm = *(const v16h*)wr;
        acc[ct] = __builtin_amdgcn_wmma_f32_16x16x32_f16(false, a, false, bm, (short)0, acc[ct], false, false);
      }
    }
  }
#pragma unroll
  for (int ct = 0; ct < 8; ++ct) {
    if (ct < nct) {
      int ch = ct*16 + row;
      float bv = bias ? bias[ch] : 0.f;
#pragma unroll
      for (int r = 0; r < 8; ++r) {
        int orow = r0 + r + (half ? 8 : 0);
        if (orow < rows) {
          float v = acc[ct][r] + bv;
          Y[(size_t)orow * Cout + ch] = act ? leakyf(v) : v;
        }
      }
    }
  }
}

// ---------------------------------------------------------------------------
// three-NN (3 nearest known for each unknown), squared dists clamped >= 0.
// ---------------------------------------------------------------------------
__global__ void three_nn_kernel(const float* __restrict__ unk, const float* __restrict__ kn,
                                int Mu, int Mk, int* __restrict__ oidx, float* __restrict__ odist) {
  int b = blockIdx.y;
  const float* U = unk + (size_t)b * Mu * 3;
  const float* Q = kn + (size_t)b * Mk * 3;
  int q = blockIdx.x * blockDim.x + threadIdx.x;
  __shared__ float sx[256], sy[256], sz[256];
  float qx = 0.f, qy = 0.f, qz = 0.f;
  if (q < Mu) { qx = U[q*3+0]; qy = U[q*3+1]; qz = U[q*3+2]; }
  float bd[3] = {3.0e38f, 3.0e38f, 3.0e38f};
  int   bi[3] = {0, 0, 0};
  for (int base = 0; base < Mk; base += 256) {
    int c = base + (int)threadIdx.x;
    if (c < Mk) { sx[threadIdx.x] = Q[c*3+0]; sy[threadIdx.x] = Q[c*3+1]; sz[threadIdx.x] = Q[c*3+2]; }
    int pf = c + 256;
    if (pf < Mk) __builtin_prefetch(&Q[(size_t)pf * 3], 0, 0);
    __syncthreads();
    int lim = Mk - base; if (lim > 256) lim = 256;
    if (q < Mu) {
      for (int j = 0; j < lim; ++j) {
        float dx = qx - sx[j], dy = qy - sy[j], dz = qz - sz[j];
        float d = dx*dx + dy*dy + dz*dz;
        if (d < bd[2]) {
          float cd = d; int ci = base + j;
#pragma unroll
          for (int t = 0; t < 3; ++t) {
            if (cd < bd[t]) { float td = bd[t]; int ti = bi[t]; bd[t] = cd; bi[t] = ci; cd = td; ci = ti; }
          }
        }
      }
    }
    __syncthreads();
  }
  if (q < Mu) {
#pragma unroll
    for (int t = 0; t < 3; ++t) {
      oidx [((size_t)b*Mu + q)*3 + t] = bi[t];
      odist[((size_t)b*Mu + q)*3 + t] = bd[t] < 0.f ? 0.f : bd[t];
    }
  }
}

// ---------------------------------------------------------------------------
// cat[b,m,:] = [ three_interp(featk, nn), feats[b,m,:] ]
// ---------------------------------------------------------------------------
__global__ void interp_concat_kernel(const float* __restrict__ featk, const int* __restrict__ nnidx,
                                     const float* __restrict__ nndist, const float* __restrict__ feats,
                                     float* __restrict__ cat, int Mu, int Mk, int C, int Cs) {
  int b = blockIdx.y;
  int Ctot = C + Cs;
  int i = blockIdx.x * blockDim.x + threadIdx.x;
  if (i >= Mu * Ctot) return;
  int m = i / Ctot, c = i - m * Ctot;
  float v;
  if (c < C) {
    const int*   id = nnidx  + ((size_t)b*Mu + m) * 3;
    const float* dd = nndist + ((size_t)b*Mu + m) * 3;
    float r0 = 1.f / (dd[0] + 1e-8f);
    float r1 = 1.f / (dd[1] + 1e-8f);
    float r2 = 1.f / (dd[2] + 1e-8f);
    float inv = 1.f / (r0 + r1 + r2);
    const float* Fk = featk + (size_t)b * Mk * C;
    v = (r0 * Fk[(size_t)id[0]*C + c] + r1 * Fk[(size_t)id[1]*C + c] + r2 * Fk[(size_t)id[2]*C + c]) * inv;
  } else {
    v = feats[((size_t)b*Mu + m) * Cs + (c - C)];
  }
  cat[((size_t)b*Mu + m) * Ctot + c] = v;
}

// ---------------------------------------------------------------------------
// Orchestration
// ---------------------------------------------------------------------------
extern "C" void kernel_launch(void* const* d_in, const int* in_sizes, int n_in,
                              void* d_out, int out_size, void* d_ws, size_t ws_size,
                              hipStream_t stream) {
  (void)in_sizes; (void)n_in; (void)out_size; (void)ws_size;
  const float* x          = (const float*)d_in[0];
  const float* pos0       = (const float*)d_in[1];
  const float* w_self0    = (const float*)d_in[2];
  const float* w_edge0    = (const float*)d_in[3];
  const float* w_self1    = (const float*)d_in[4];
  const float* w_edge1    = (const float*)d_in[5];
  const float* w_self2    = (const float*)d_in[6];
  const float* w_edge2    = (const float*)d_in[7];
  const float* w_up1      = (const float*)d_in[8];
  const float* b_up1      = (const float*)d_in[9];
  const float* w_uc1_self = (const float*)d_in[10];
  const float* w_uc1_edge = (const float*)d_in[11];
  const float* w_up0      = (const float*)d_in[12];
  const float* b_up0      = (const float*)d_in[13];
  const float* w_uc0_self = (const float*)d_in[14];
  const float* w_uc0_edge = (const float*)d_in[15];
  const float* w_out      = (const float*)d_in[16];
  const float* b_out      = (const float*)d_in[17];

  const int B = 2, N = 8192, N1 = 4096, N2 = 2048;

  char* p = (char*)d_ws;
  auto carve = [&](size_t bytes) -> char* {
    char* r = p; p += (bytes + 255) & ~(size_t)255; return r;
  };

  int*   idx0     = (int*)  carve((size_t)B*N*16*4);
  float* feat0    = (float*)carve((size_t)B*N*64*4);
  int*   fidx1    = (int*)  carve((size_t)B*N1*4);
  float* pos1     = (float*)carve((size_t)B*N1*3*4);
  float* f1in     = (float*)carve((size_t)B*N1*64*4);
  int*   idx1     = (int*)  carve((size_t)B*N1*16*4);
  float* feat1    = (float*)carve((size_t)B*N1*128*4);
  int*   fidx2    = (int*)  carve((size_t)B*N2*4);
  float* pos2     = (float*)carve((size_t)B*N2*3*4);
  float* f2in     = (float*)carve((size_t)B*N2*128*4);
  int*   idx2k    = (int*)  carve((size_t)B*N2*16*4);
  float* feat2    = (float*)carve((size_t)B*N2*256*4);
  int*   nnidxA   = (int*)  carve((size_t)B*N1*3*4);
  float* nndistA  = (float*)carve((size_t)B*N1*3*4);
  float* cat1     = (float*)carve((size_t)B*N1*384*4);
  float* feat_up1 = (float*)carve((size_t)B*N1*128*4);
  float* fup1c    = (float*)carve((size_t)B*N1*128*4);
  int*   nnidxB   = (int*)  carve((size_t)B*N*3*4);
  float* nndistB  = (float*)carve((size_t)B*N*3*4);
  float* cat0     = (float*)carve((size_t)B*N*192*4);
  float* feat_up0 = (float*)carve((size_t)B*N*64*4);
  float* feat_ob  = (float*)carve((size_t)B*N*64*4);
  _Float16* wt_e0   = (_Float16*)carve((size_t)64 *32 *2);
  _Float16* wt_e1   = (_Float16*)carve((size_t)128*96 *2);
  _Float16* wt_e2   = (_Float16*)carve((size_t)256*160*2);
  _Float16* wt_up1  = (_Float16*)carve((size_t)128*384*2);
  _Float16* wt_uc1e = (_Float16*)carve((size_t)128*160*2);
  _Float16* wt_up0  = (_Float16*)carve((size_t)64 *192*2);
  _Float16* wt_uc0e = (_Float16*)carve((size_t)64 *96 *2);
  _Float16* wt_o    = (_Float16*)carve((size_t)128*64 *2);

  auto cvt = [&](const float* w, _Float16* wt, int cin, int cout, int kpad) {
    int tot = cout * kpad;
    wt_convert_kernel<<<dim3((tot + 255) / 256), 256, 0, stream>>>(w, wt, cin, cout, kpad);
  };
  cvt(w_edge0,    wt_e0,   8,   64,  32);
  cvt(w_edge1,    wt_e1,   68,  128, 96);
  cvt(w_edge2,    wt_e2,   132, 256, 160);
  cvt(w_up1,      wt_up1,  384, 128, 384);
  cvt(w_uc1_edge, wt_uc1e, 132, 128, 160);
  cvt(w_up0,      wt_up0,  192, 64,  192);
  cvt(w_uc0_edge, wt_uc0e, 68,  64,  96);
  cvt(w_out,      wt_o,    64,  128, 64);

  auto egrid = [](int M) { return dim3((unsigned)((M + WPB - 1) / WPB), 2); };
  auto lgrid = [](int rows) { return dim3((unsigned)((rows + 16*WPB - 1) / (16*WPB))); };

  // ---- level 0 edge conv ----
  knn16_kernel<<<dim3((N + 255) / 256, B), 256, 0, stream>>>(pos0, idx0, N);
  edgeconv_wmma_kernel<<<egrid(N), 32*WPB, 0, stream>>>(x, pos0, idx0, w_self0, wt_e0, feat0, N, 4, 64, 32);

  // ---- downsample to N1, edge conv 1 ----
  fps_kernel<<<dim3(B), FPS_T, 0, stream>>>(pos0, fidx1, N, N1);
  gather_kernel<<<dim3((N1*3  + 255) / 256, B), 256, 0, stream>>>(pos0,  fidx1, pos1, N, N1, 3);
  gather_kernel<<<dim3((N1*64 + 255) / 256, B), 256, 0, stream>>>(feat0, fidx1, f1in, N, N1, 64);
  knn16_kernel<<<dim3((N1 + 255) / 256, B), 256, 0, stream>>>(pos1, idx1, N1);
  edgeconv_wmma_kernel<<<egrid(N1), 32*WPB, 0, stream>>>(f1in, pos1, idx1, w_self1, wt_e1, feat1, N1, 64, 128, 96);

  // ---- downsample to N2, edge conv 2 ----
  fps_kernel<<<dim3(B), FPS_T, 0, stream>>>(pos1, fidx2, N1, N2);
  gather_kernel<<<dim3((N2*3   + 255) / 256, B), 256, 0, stream>>>(pos1,  fidx2, pos2, N1, N2, 3);
  gather_kernel<<<dim3((N2*128 + 255) / 256, B), 256, 0, stream>>>(feat1, fidx2, f2in, N1, N2, 128);
  knn16_kernel<<<dim3((N2 + 255) / 256, B), 256, 0, stream>>>(pos2, idx2k, N2);
  edgeconv_wmma_kernel<<<egrid(N2), 32*WPB, 0, stream>>>(f2in, pos2, idx2k, w_self2, wt_e2, feat2, N2, 128, 256, 160);

  // ---- up path: level 2 -> 1 ----
  three_nn_kernel<<<dim3((N1 + 255) / 256, B), 256, 0, stream>>>(pos1, pos2, N1, N2, nnidxA, nndistA);
  interp_concat_kernel<<<dim3((N1*384 + 255) / 256, B), 256, 0, stream>>>(feat2, nnidxA, nndistA, feat1, cat1, N1, N2, 256, 128);
  linear_wmma_kernel<<<lgrid(B*N1), 32*WPB, 0, stream>>>(cat1, wt_up1, b_up1, feat_up1, B*N1, 384, 128, 384, 1);
  edgeconv_wmma_kernel<<<egrid(N1), 32*WPB, 0, stream>>>(feat_up1, pos1, idx1, w_uc1_self, wt_uc1e, fup1c, N1, 128, 128, 160);

  // ---- up path: level 1 -> 0 ----
  three_nn_kernel<<<dim3((N + 255) / 256, B), 256, 0, stream>>>(pos0, pos1, N, N1, nnidxB, nndistB);
  interp_concat_kernel<<<dim3((N*192 + 255) / 256, B), 256, 0, stream>>>(fup1c, nnidxB, nndistB, feat0, cat0, N, N1, 128, 64);
  linear_wmma_kernel<<<lgrid(B*N), 32*WPB, 0, stream>>>(cat0, wt_up0, b_up0, feat_up0, B*N, 192, 64, 192, 1);
  edgeconv_wmma_kernel<<<egrid(N), 32*WPB, 0, stream>>>(feat_up0, pos0, idx0, w_uc0_self, wt_uc0e, feat_ob, N, 64, 64, 96);

  // ---- output projection (no activation) ----
  linear_wmma_kernel<<<lgrid(B*N), 32*WPB, 0, stream>>>(feat_ob, wt_o, b_out, (float*)d_out, B*N, 64, 128, 64, 0);
}
